// SNNLayer_20246475834024
// MI455X (gfx1250) — matrix operand order repaired and checked by reading, not compiled, not run on hardware
//
#include <hip/hip_runtime.h>
#include <hip/hip_bf16.h>
#include <hip/hip_fp16.h>

// ---------------------------------------------------------------------------
// SNN layer for MI455X (gfx1250, wave32, WMMA).
//   Stage 1: y = x @ rnn_kernel           (f16 WMMA GEMM, 262144x128 @ 128x128)
//   Stage 2: proj = conv_t(y) + ubias     (32-tap shared kernel, VALU/bandwidth)
//   Stage 3: sequential scan, WMMA per step, resident-register R fragments,
//            double-buffered LDS state (1 barrier/step), prefetched proj.
// conv and projection commute (shared tap kernel, both linear), so the big
// GEMM runs first and the conv operates on the 128-wide projected space.
// ---------------------------------------------------------------------------

typedef __attribute__((ext_vector_type(16))) _Float16 v16h;
typedef __attribute__((ext_vector_type(8)))  _Float16 v8h;
typedef __attribute__((ext_vector_type(4)))  _Float16 v4h;
typedef __attribute__((ext_vector_type(8)))  float    v8f;
typedef __attribute__((ext_vector_type(4)))  float    v4f;

#define B_  256
#define L_  1024
#define W_  128
#define P_  32
#define T_  993     // L - P + 1
#define U_  128

// ---------------------------------------------------------------------------
// Pre-swizzle both 128x128 weight matrices into WMMA B-operand fragments.
// Fragment f = kb*8 + nb covers K rows [kb*32, kb*32+32), N cols [nb*16, nb*16+16).
// ISA B layout (16-bit, 32x16): lane n (0..15) holds K={0..7,16..23} as elements 0..15;
// lane n+16 holds K={8..15,24..31}. Stored so each lane's v16h is 32 contiguous bytes.
// Also computes ubias[u] = psc_bias * sum_w K[w][u] + rnn_bias[u].
// ---------------------------------------------------------------------------
__global__ void snn_prep_kernel(const float* __restrict__ K,
                                const float* __restrict__ R,
                                const float* __restrict__ pscb,
                                const float* __restrict__ rnnb,
                                _Float16* __restrict__ Kf,
                                _Float16* __restrict__ Rf,
                                float* __restrict__ ubias) {
    const int lane = threadIdx.x & 31;
    const int wv   = threadIdx.x >> 5;          // 0 -> K, 1 -> R
    const float*    Wsrc = (wv == 0) ? K  : R;
    _Float16*       Wdst = (wv == 0) ? Kf : Rf;
    const int n  = lane & 15;
    const int kh = lane >> 4;                   // K half: 0 -> K 0..7,16..23; 1 -> K 8..15,24..31
    for (int f = 0; f < 32; ++f) {
        const int kb = f >> 3;
        const int nb = f & 7;
        #pragma unroll
        for (int i = 0; i < 16; ++i) {
            const int k = kb * 32 + kh * 8 + (i & 7) + (i >> 3) * 16;
            Wdst[f * 512 + lane * 16 + i] = (_Float16)Wsrc[k * U_ + nb * 16 + n];
        }
    }
    // per-unit fused bias
    const float pb = pscb[0];
    for (int u = threadIdx.x; u < U_; u += 64) {
        float s = 0.f;
        for (int w = 0; w < W_; ++w) s += K[w * U_ + u];
        ubias[u] = pb * s + rnnb[u];
    }
}

// ---------------------------------------------------------------------------
// Stage 1: y(f16) = x(f32) @ rnn_kernel.  One wave per 16-row M tile.
// A layout (16-bit, 16x32): lane<16 -> elems {K 0..7, 16..23}; lane>=16 ->
// {8..15, 24..31}: two contiguous 8-float chunks per fragment.
// x is streamed read-once -> nontemporal loads keep L2 free for y/proj reuse.
// ---------------------------------------------------------------------------
__global__ void snn_gemm_xk_kernel(const float* __restrict__ x,
                                   const _Float16* __restrict__ Kf,
                                   _Float16* __restrict__ y) {
    const int wave = blockIdx.x * (blockDim.x >> 5) + (threadIdx.x >> 5);
    const int lane = threadIdx.x & 31;
    const int m    = lane & 15;
    const int hi   = lane >> 4;

    const float* rowp = x + (size_t)(wave * 16 + m) * W_;

    v16h a[4];
    #pragma unroll
    for (int kb = 0; kb < 4; ++kb) {
        const int base = kb * 32 + hi * 8;
        v4f f0 = __builtin_nontemporal_load((const v4f*)(rowp + base));
        v4f f1 = __builtin_nontemporal_load((const v4f*)(rowp + base + 4));
        v4f f2 = __builtin_nontemporal_load((const v4f*)(rowp + base + 16));
        v4f f3 = __builtin_nontemporal_load((const v4f*)(rowp + base + 20));
        #pragma unroll
        for (int j = 0; j < 4; ++j) {
            a[kb][j]      = (_Float16)f0[j];
            a[kb][4 + j]  = (_Float16)f1[j];
            a[kb][8 + j]  = (_Float16)f2[j];
            a[kb][12 + j] = (_Float16)f3[j];
        }
    }

    #pragma unroll
    for (int nb = 0; nb < 8; ++nb) {
        v8f c = {};
        #pragma unroll
        for (int kb = 0; kb < 4; ++kb) {
            v16h bf = *(const v16h*)(Kf + (kb * 8 + nb) * 512 + lane * 16);
            c = __builtin_amdgcn_wmma_f32_16x16x32_f16(
                    false, a[kb], false, bf, (short)0, c, false, false);
        }
        // C layout: lane l -> col = l%16, rows v + 8*(l/16)
        #pragma unroll
        for (int v = 0; v < 8; ++v) {
            y[(size_t)(wave * 16 + v + 8 * hi) * U_ + nb * 16 + m] = (_Float16)c[v];
        }
    }
}

// ---------------------------------------------------------------------------
// Stage 2: proj[b,t,u] = ubias[u] + sum_p k[p] * y[b, t+p, u].
// Each thread produces 8 consecutive t for one float4 of u (4.9x read reuse;
// y is L2-resident at 64 MB).
// ---------------------------------------------------------------------------
__global__ void snn_conv_kernel(const _Float16* __restrict__ y,
                                const float* __restrict__ pk,
                                const float* __restrict__ ubias,
                                float* __restrict__ proj) {
    const int gid  = blockIdx.x * blockDim.x + threadIdx.x;
    const int u4   = gid & 31;
    const int rest = gid >> 5;
    const int tb   = rest % 125;
    const int b    = rest / 125;
    if (b >= B_) return;
    const int t0 = tb * 8;

    float kk[P_];
    #pragma unroll
    for (int p = 0; p < P_; ++p) kk[p] = pk[p];

    v4f bias4 = *(const v4f*)(ubias + u4 * 4);
    v4f acc[8];
    #pragma unroll
    for (int j = 0; j < 8; ++j) acc[j] = bias4;

    const _Float16* ybase = y + (size_t)b * L_ * U_ + u4 * 4;
    #pragma unroll
    for (int i = 0; i < 39; ++i) {
        const int row = t0 + i;
        v4f yv = {0.f, 0.f, 0.f, 0.f};
        if (row < L_) {
            v4h h4 = *(const v4h*)(ybase + (size_t)row * U_);
            yv[0] = (float)h4[0]; yv[1] = (float)h4[1];
            yv[2] = (float)h4[2]; yv[3] = (float)h4[3];
        }
        #pragma unroll
        for (int j = 0; j < 8; ++j) {
            const int p = i - j;                 // constant after full unroll
            if (p >= 0 && p < P_) acc[j] += kk[p] * yv;
        }
    }
    #pragma unroll
    for (int j = 0; j < 8; ++j) {
        const int t = t0 + j;
        if (t < T_)
            *(v4f*)(proj + ((size_t)b * T_ + t) * U_ + u4 * 4) = acc[j];
    }
}

// ---------------------------------------------------------------------------
// Stage 3: sequential scan. 16 blocks x 256 threads; block owns 16 batches,
// wave nb owns 16 output units and keeps its 4 R fragments resident in
// registers for all 993 steps.  h state is double-buffered f16 in LDS in
// plain [m][u] order (A fragment = two ds_load_b128), giving exactly one
// barrier per step.  proj for step t+1 is prefetched (nontemporal) at the
// top of step t so its L2 latency hides behind the wmma chain + barrier.
// ---------------------------------------------------------------------------
__global__ void snn_scan_kernel(const float* __restrict__ proj,
                                const _Float16* __restrict__ Rf,
                                float* __restrict__ out) {
    __shared__ __align__(16) _Float16 hsh[2][16 * U_];

    const int nb    = threadIdx.x >> 5;
    const int lane  = threadIdx.x & 31;
    const int m     = lane & 15;
    const int hi    = lane >> 4;
    const int btile = blockIdx.x * 16;
    const int col   = nb * 16 + m;

    for (int i = threadIdx.x; i < 16 * U_; i += blockDim.x) {
        hsh[0][i] = (_Float16)0.f;
        hsh[1][i] = (_Float16)0.f;
    }

    v16h r[4];
    #pragma unroll
    for (int kb = 0; kb < 4; ++kb)
        r[kb] = *(const v16h*)(Rf + (kb * 8 + nb) * 512 + lane * 16);

    // per-row base pointers (strength-reduced address math off the hot path)
    const float* pbase[8];
    float*       obase[8];
    #pragma unroll
    for (int v = 0; v < 8; ++v) {
        const int row = v + 8 * hi;
        pbase[v] = proj + (size_t)(btile + row) * T_ * U_ + col;
        obase[v] = out  + (size_t)(btile + row) * ((size_t)T_ * U_) + col;
    }

    __syncthreads();

    float pc[8];
    #pragma unroll
    for (int v = 0; v < 8; ++v)
        pc[v] = __builtin_nontemporal_load(pbase[v]);        // t = 0

    const int mbase = m * U_;
    for (int t = 0; t < T_; ++t) {
        // prefetch proj for t+1 (independent of h; hides L2 latency)
        float pn[8];
        #pragma unroll
        for (int v = 0; v < 8; ++v)
            pn[v] = (t + 1 < T_)
                  ? __builtin_nontemporal_load(pbase[v] + (size_t)(t + 1) * U_)
                  : 0.f;

        // A fragments of h (previous step) from read buffer
        const _Float16* hb = &hsh[t & 1][0];
        v16h a[4];
        #pragma unroll
        for (int kb = 0; kb < 4; ++kb) {
            const int base = mbase + kb * 32 + hi * 8;
            v8h lo  = *(const v8h*)(hb + base);
            v8h hi8 = *(const v8h*)(hb + base + 16);
            #pragma unroll
            for (int j = 0; j < 8; ++j) { a[kb][j] = lo[j]; a[kb][8 + j] = hi8[j]; }
        }

        // out_t = proj_t + h @ R
        v8f c;
        #pragma unroll
        for (int v = 0; v < 8; ++v) c[v] = pc[v];
        #pragma unroll
        for (int kb = 0; kb < 4; ++kb)
            c = __builtin_amdgcn_wmma_f32_16x16x32_f16(
                    false, a[kb], false, r[kb], (short)0, c, false, false);

        // sigmoid gate, emit n (streaming store), next state into write buffer
        _Float16* hw = &hsh[(t + 1) & 1][0];
        #pragma unroll
        for (int v = 0; v < 8; ++v) {
            const float o = c[v];
            const float n = 1.0f / (1.0f + __expf(-o));
            __builtin_nontemporal_store(n, obase[v] + (size_t)t * U_);
            hw[(v + 8 * hi) * U_ + col] = (_Float16)(o * (1.0f - n));
        }

        __syncthreads();            // new state visible before next step's reads

        #pragma unroll
        for (int v = 0; v < 8; ++v) pc[v] = pn[v];
    }
}

// ---------------------------------------------------------------------------
extern "C" void kernel_launch(void* const* d_in, const int* in_sizes, int n_in,
                              void* d_out, int out_size, void* d_ws, size_t ws_size,
                              hipStream_t stream) {
    const float* x    = (const float*)d_in[0];   // (256,1024,128)
    const float* psck = (const float*)d_in[1];   // (32,)
    const float* pscb = (const float*)d_in[2];   // (1,)
    const float* K    = (const float*)d_in[3];   // (128,128)
    const float* R    = (const float*)d_in[4];   // (128,128)
    const float* rb   = (const float*)d_in[5];   // (128,)
    float* out = (float*)d_out;                  // (256, 993*128)

    char* ws = (char*)d_ws;
    _Float16* Kf    = (_Float16*)(ws);                                   // 32 KB
    _Float16* Rf    = (_Float16*)(ws + 32 * 1024);                       // 32 KB
    float*    ubias = (float*)   (ws + 64 * 1024);                       // 512 B
    _Float16* y     = (_Float16*)(ws + 68 * 1024);                       // 64 MB
    float*    proj  = (float*)   (ws + 68 * 1024 +
                                  (size_t)B_ * L_ * U_ * sizeof(_Float16)); // ~124 MB

    snn_prep_kernel<<<1, 64, 0, stream>>>(K, R, pscb, rb, Kf, Rf, ubias);

    // 262144 rows / 16 per wave = 16384 waves; 8 waves per block.
    snn_gemm_xk_kernel<<<2048, 256, 0, stream>>>(x, Kf, y);

    // 256 batches * 125 t-blocks * 32 u4-groups = 1,024,000 threads.
    snn_conv_kernel<<<4000, 256, 0, stream>>>(y, psck, ubias, proj);

    snn_scan_kernel<<<B_ / 16, 256, 0, stream>>>(proj, Rf, out);
}
